// GraphEncoder_6760278524117
// MI455X (gfx1250) — matrix operand order repaired and checked by reading, not compiled, not run on hardware
//
#include <hip/hip_runtime.h>
#include <hip/hip_bf16.h>

// ---------------------------------------------------------------------------
// GAT forward for MI455X (gfx1250, wave32).
// GEMMs: v_wmma_f32_16x16x32_bf16 with the B panel staged in LDS in WMMA
// B-layout (one contiguous 32B ds_load per fragment). Edge softmax/scatter is
// HBM-bound: coalesced float4 gathers + f32 global atomics.
// ---------------------------------------------------------------------------

typedef __attribute__((ext_vector_type(16))) __bf16 v16bf;
typedef __attribute__((ext_vector_type(8)))  float  v8f;

#define HEADS 4
#define DH    128
#define NEG_SLOPE 0.2f

#define BM 128            // block M rows (8 waves x 16)
#define BN 64             // block N cols (4 wmma tiles per wave)
#define BSTRIDE 144       // LDS row stride in bf16 elems (128 K + 16 pad)

// ---- monotonic float<->uint mapping so atomicMax(u32) == float max ----------
__device__ __forceinline__ unsigned encF(float f) {
    unsigned u = __float_as_uint(f);
    return (u & 0x80000000u) ? ~u : (u | 0x80000000u);
}
__device__ __forceinline__ float decF(unsigned u) {
    return (u & 0x80000000u) ? __uint_as_float(u & 0x7FFFFFFFu)
                             : __uint_as_float(~u);
}

// ---------------------------------------------------------------------------
// WMMA GEMM: C[M,Nc] = A[M,K] * B[K,Nc] (+bias). fp32 in/out, bf16 compute.
// Block: 256 threads = 8 waves; computes a BM x BN tile. B panel (K x BN)
// is cooperatively transposed+converted into LDS as Bt[col][k] so each
// wave B-fragment is one contiguous 32-byte LDS read.
// A lane layout (16-bit A 16x32, ISA 7.12.2): row = lane&15,
//   kbase = (lane>=16)?16:0, elems 0..15 -> K=kbase..+15 (contiguous; ISA table order differs)
// B lane layout: col = lane&15, elem e -> K = (lane>=16?16:0)+e (contiguous, same order as A)
// C/D: vgpr v -> row v + 8*(lane>=16), col = lane&15.
// K must be a multiple of 32 (here K=128).
// ---------------------------------------------------------------------------
__global__ __launch_bounds__(256)
void wmma_gemm_bf16(const float* __restrict__ A,
                    const float* __restrict__ B,
                    const float* __restrict__ bias,
                    float* __restrict__ C,
                    int M, int K, int Nc) {
    __shared__ __bf16 Bt[BN * BSTRIDE];

    const int tilesN = Nc / BN;
    const int bM = blockIdx.x / tilesN;
    const int bN = blockIdx.x % tilesN;

    // ---- stage B panel: Bt[c][k] = bf16(B[k][bN*BN + c]) (coalesced in c) --
    for (int idx = threadIdx.x; idx < BN * K; idx += 256) {
        const int k = idx / BN;
        const int c = idx % BN;
        Bt[c * BSTRIDE + k] = (__bf16)B[(size_t)k * Nc + bN * BN + c];
    }
    __syncthreads();

    const int wave = threadIdx.x >> 5;
    const int lane = threadIdx.x & 31;
    const int half = lane >> 4;
    const int l15  = lane & 15;

    const int r0 = bM * BM + wave * 16;            // wave's first output row
    if (r0 >= M) return;                           // wave-uniform exit

    int rowA = r0 + l15;
    if (rowA >= M) rowA = M - 1;                   // clamp (stores guarded)

    v8f acc0 = {}, acc1 = {}, acc2 = {}, acc3 = {};
    for (int k0 = 0; k0 < K; k0 += 32) {
        const int kbA = k0 + half * 16;
        const float* ap = A + (size_t)rowA * K + kbA;
        v16bf a;
#pragma unroll
        for (int i = 0; i < 8; ++i) {
            a[i]     = (__bf16)ap[i];              // K = kbA + i
            a[i + 8] = (__bf16)ap[i + 8];          // K = kbA + 8 + i
        }
        const int kbB = k0 + half * 16;
        const v16bf b0 = *(const v16bf*)&Bt[( 0 + l15) * BSTRIDE + kbB];
        const v16bf b1 = *(const v16bf*)&Bt[(16 + l15) * BSTRIDE + kbB];
        const v16bf b2 = *(const v16bf*)&Bt[(32 + l15) * BSTRIDE + kbB];
        const v16bf b3 = *(const v16bf*)&Bt[(48 + l15) * BSTRIDE + kbB];
        acc0 = __builtin_amdgcn_wmma_f32_16x16x32_bf16(false, a, false, b0, (short)0, acc0, false, false);
        acc1 = __builtin_amdgcn_wmma_f32_16x16x32_bf16(false, a, false, b1, (short)0, acc1, false, false);
        acc2 = __builtin_amdgcn_wmma_f32_16x16x32_bf16(false, a, false, b2, (short)0, acc2, false, false);
        acc3 = __builtin_amdgcn_wmma_f32_16x16x32_bf16(false, a, false, b3, (short)0, acc3, false, false);
    }

    // ---- epilogue ----------------------------------------------------------
    const size_t nstride = (size_t)Nc;
    const int    cbase   = bN * BN + l15;
    float bv0 = 0.f, bv1 = 0.f, bv2 = 0.f, bv3 = 0.f;
    if (bias) {
        bv0 = bias[cbase];      bv1 = bias[cbase + 16];
        bv2 = bias[cbase + 32]; bv3 = bias[cbase + 48];
    }
    const v8f* accs[4] = { &acc0, &acc1, &acc2, &acc3 };
    const float bvs[4] = { bv0, bv1, bv2, bv3 };

    if (r0 + 16 <= M) {                            // full tile: unguarded
        float* cp = C + (size_t)(r0 + 8 * half) * nstride + cbase;
#pragma unroll
        for (int v = 0; v < 8; ++v) {
            float* row = cp + (size_t)v * nstride;
#pragma unroll
            for (int nt = 0; nt < 4; ++nt)
                row[nt * 16] = (*accs[nt])[v] + bvs[nt];
        }
    } else {                                       // edge tile: guarded
#pragma unroll
        for (int v = 0; v < 8; ++v) {
            const int r = r0 + v + 8 * half;
            if (r < M) {
                float* row = C + (size_t)r * nstride + cbase;
#pragma unroll
                for (int nt = 0; nt < 4; ++nt)
                    row[nt * 16] = (*accs[nt])[v] + bvs[nt];
            }
        }
    }
}

// ---------------------------------------------------------------------------
// al_src[n,h] = dot(xh[n,h,:], a_src[h,:]);  same for al_dst. 1 thread/(n,h).
// ---------------------------------------------------------------------------
__global__ void attn_coeff(const float* __restrict__ xh,
                           const float* __restrict__ a_src,
                           const float* __restrict__ a_dst,
                           float* __restrict__ al_src,
                           float* __restrict__ al_dst, int N) {
    const int t = blockIdx.x * blockDim.x + threadIdx.x;
    const int n = t >> 2, h = t & 3;
    if (n >= N) return;
    const float* xp = xh + (size_t)n * (HEADS * DH) + h * DH;
    const float* as = a_src + h * DH;
    const float* ad = a_dst + h * DH;
    float ss = 0.f, sd = 0.f;
#pragma unroll 4
    for (int d = 0; d < DH; ++d) {
        const float x = xp[d];
        ss += x * as[d];
        sd += x * ad[d];
    }
    al_src[n * HEADS + h] = ss;
    al_dst[n * HEADS + h] = sd;
}

// ---------------------------------------------------------------------------
// Pass 1: e = leaky_relu(al_src[s]+al_dst[d]); store logit; atomic max per dst
// ---------------------------------------------------------------------------
__global__ void edge_logit_max(const int* __restrict__ src,
                               const int* __restrict__ dst,
                               const float* __restrict__ al_src,
                               const float* __restrict__ al_dst,
                               float* __restrict__ elog,
                               unsigned* __restrict__ mmax,
                               int E, int Etot) {
    const int e = blockIdx.x * blockDim.x + threadIdx.x;
    if (e >= Etot) return;
    const int s = (e < E) ? src[e] : (e - E);      // self-loops appended
    const int d = (e < E) ? dst[e] : (e - E);
#pragma unroll
    for (int h = 0; h < HEADS; ++h) {
        float v = al_src[s * HEADS + h] + al_dst[d * HEADS + h];
        v = (v > 0.f) ? v : NEG_SLOPE * v;
        elog[(size_t)e * HEADS + h] = v;
        atomicMax(&mmax[d * HEADS + h], encF(v));
    }
}

// ---------------------------------------------------------------------------
// Pass 2: ex = exp(e - m[dst]); overwrite logit slot; atomic-sum per dst
// ---------------------------------------------------------------------------
__global__ void edge_expsum(const int* __restrict__ src,
                            const int* __restrict__ dst,
                            float* __restrict__ elog,      // in: logit, out: ex
                            const unsigned* __restrict__ mmax,
                            float* __restrict__ ssum,
                            int E, int Etot) {
    const int e = blockIdx.x * blockDim.x + threadIdx.x;
    if (e >= Etot) return;
    const int d = (e < E) ? dst[e] : (e - E);
#pragma unroll
    for (int h = 0; h < HEADS; ++h) {
        const float m  = decF(mmax[d * HEADS + h]);
        const float ex = __expf(elog[(size_t)e * HEADS + h] - m);
        elog[(size_t)e * HEADS + h] = ex;
        unsafeAtomicAdd(&ssum[d * HEADS + h], ex);
    }
}

// ---------------------------------------------------------------------------
// Pass 3: acc[dst,h,:] += xh[src,h,:] * alpha. One wave per (edge, head):
// 32 lanes x float4 = 128 features, coalesced gather + f32 atomic scatter.
// ---------------------------------------------------------------------------
__global__ void edge_scatter(const int* __restrict__ src,
                             const int* __restrict__ dst,
                             const float* __restrict__ ex,
                             const float* __restrict__ ssum,
                             const float* __restrict__ xh,
                             float* __restrict__ acc,
                             int E, int Etot) {
    const int gid  = blockIdx.x * blockDim.x + threadIdx.x;
    const int wid  = gid >> 5;
    const int lane = threadIdx.x & 31;
    const int e    = wid >> 2;                     // wave-uniform
    const int h    = wid & 3;
    if (e >= Etot) return;
    const int s = (e < E) ? src[e] : (e - E);
    const int d = (e < E) ? dst[e] : (e - E);
    const float alpha = ex[(size_t)e * HEADS + h] /
                        (ssum[d * HEADS + h] + 1e-16f);
    const float4 v = ((const float4*)(xh + (size_t)s * (HEADS * DH) + h * DH))[lane];
    float* ap = acc + (size_t)d * (HEADS * DH) + h * DH + lane * 4;
    unsafeAtomicAdd(ap + 0, v.x * alpha);
    unsafeAtomicAdd(ap + 1, v.y * alpha);
    unsafeAtomicAdd(ap + 2, v.z * alpha);
    unsafeAtomicAdd(ap + 3, v.w * alpha);
}

// ---------------------------------------------------------------------------
// h_next[n,d] = relu(mean_h(acc[n,h,d]) + b[d])
// ---------------------------------------------------------------------------
__global__ void finalize_layer(const float* __restrict__ acc,
                               const float* __restrict__ b,
                               float* __restrict__ hout, int N) {
    const int t = blockIdx.x * blockDim.x + threadIdx.x;
    const int n = t >> 7, d = t & 127;
    if (n >= N) return;
    const float* ap = acc + (size_t)n * (HEADS * DH) + d;
    float v = 0.25f * (ap[0] + ap[DH] + ap[2 * DH] + ap[3 * DH]) + b[d];
    hout[(size_t)n * DH + d] = fmaxf(v, 0.0f);
}

// ---------------------------------------------------------------------------
// Host-side orchestration
// ---------------------------------------------------------------------------
extern "C" void kernel_launch(void* const* d_in, const int* in_sizes, int n_in,
                              void* d_out, int out_size, void* d_ws, size_t ws_size,
                              hipStream_t stream) {
    const float* x  = (const float*)d_in[0];
    const int*   ei = (const int*)d_in[1];
    const int N    = in_sizes[0] / DH;      // 50000
    const int E    = in_sizes[1] / 2;       // 400000
    const int Etot = E + N;                 // + self-loops
    const int* src = ei;
    const int* dst = ei + E;

    const float* W[3]  = { (const float*)d_in[2],  (const float*)d_in[6],  (const float*)d_in[10] };
    const float* AS[3] = { (const float*)d_in[3],  (const float*)d_in[7],  (const float*)d_in[11] };
    const float* AD[3] = { (const float*)d_in[4],  (const float*)d_in[8],  (const float*)d_in[12] };
    const float* BB[3] = { (const float*)d_in[5],  (const float*)d_in[9],  (const float*)d_in[13] };
    const float* Wp = (const float*)d_in[14];
    const float* bp = (const float*)d_in[15];

    // workspace carve-up (floats)
    float* ws = (float*)d_ws;
    size_t off = 0;
    float*    hbuf   = ws + off; off += (size_t)N * DH;          // layer output
    float*    xh     = ws + off; off += (size_t)N * HEADS * DH;  // h @ W
    float*    acc    = ws + off; off += (size_t)N * HEADS * DH;  // segment sums
    float*    al_src = ws + off; off += (size_t)N * HEADS;
    float*    al_dst = ws + off; off += (size_t)N * HEADS;
    float*    ssum   = ws + off; off += (size_t)N * HEADS;
    unsigned* mmax   = (unsigned*)(ws + off); off += (size_t)N * HEADS;
    float*    elog   = ws + off; off += (size_t)Etot * HEADS;    // logit -> ex
    (void)ws_size;

    const int THREADS = 256;
    auto cdiv = [](long long a, long long b) { return (int)((a + b - 1) / b); };

    const float* hcur = x;
    for (int l = 0; l < 3; ++l) {
        // zero per-layer accumulators (graph-capturable)
        hipMemsetAsync(acc,  0, (size_t)N * HEADS * DH * sizeof(float), stream);
        hipMemsetAsync(ssum, 0, (size_t)N * HEADS * sizeof(float), stream);
        hipMemsetAsync(mmax, 0, (size_t)N * HEADS * sizeof(unsigned), stream); // enc(-inf)~0

        // xh = hcur @ W[l]   -> [N, 512]
        {
            const int blocks = cdiv(N, BM) * ((HEADS * DH) / BN);
            wmma_gemm_bf16<<<blocks, THREADS, 0, stream>>>(
                hcur, W[l], nullptr, xh, N, DH, HEADS * DH);
        }
        attn_coeff<<<cdiv((long long)N * HEADS, THREADS), THREADS, 0, stream>>>(
            xh, AS[l], AD[l], al_src, al_dst, N);
        edge_logit_max<<<cdiv(Etot, THREADS), THREADS, 0, stream>>>(
            src, dst, al_src, al_dst, elog, mmax, E, Etot);
        edge_expsum<<<cdiv(Etot, THREADS), THREADS, 0, stream>>>(
            src, dst, elog, mmax, ssum, E, Etot);
        edge_scatter<<<cdiv((long long)Etot * HEADS * 32, THREADS), THREADS, 0, stream>>>(
            src, dst, elog, ssum, xh, acc, E, Etot);
        finalize_layer<<<cdiv((long long)N * DH, THREADS), THREADS, 0, stream>>>(
            acc, BB[l], hbuf, N);
        hcur = hbuf;
    }

    // out = hbuf @ Wp + bp  -> [N, 128]
    {
        const int blocks = cdiv(N, BM) * (DH / BN);
        wmma_gemm_bf16<<<blocks, THREADS, 0, stream>>>(
            hbuf, Wp, bp, (float*)d_out, N, DH, DH);
    }
    (void)n_in; (void)out_size;
}
